// MHDM_68968584839217
// MI455X (gfx1250) — compile-verified, matrix operand order checked by reading, not run on hardware
//
#include <hip/hip_runtime.h>
#include <hip/hip_bf16.h>

// ---------------------------------------------------------------------------
// Shapes (fixed by the reference): B=2, N=2048, D=1024, H=16, DH=64
// ---------------------------------------------------------------------------
typedef __attribute__((ext_vector_type(16))) _Float16 v16h;
typedef __attribute__((ext_vector_type(8)))  _Float16 v8h;
typedef __attribute__((ext_vector_type(8)))  float    v8f;

union V16U { v16h v; v8h h[2]; };
union V8U  { v8h v; _Float16 e[8]; };

// Load a 16-element f16 WMMA fragment slice: 8 contiguous at p, 8 at p+16.
// Matches the 16-bit A/B 16x32 VGPR layout (K = halfsel*8 + {0..7, 16..23}).
__device__ __forceinline__ v16h ld16(const _Float16* p) {
    V16U u;
    u.h[0] = *(const v8h*)(p);
    u.h[1] = *(const v8h*)(p + 16);
    return u.v;
}

__device__ __forceinline__ v8f wmma_f16(v16h a, v16h b, v8f c) {
    return __builtin_amdgcn_wmma_f32_16x16x32_f16(
        /*neg_a=*/false, a, /*neg_b=*/false, b,
        /*c_mod=*/(short)0, c, /*reuse_a=*/false, /*reuse_b=*/false);
}

// Exchange with half-partner lane (lane ^ 16) via v_permlanex16_b32: pure VALU,
// no DS round-trip / s_wait_dscnt (confirmed ds=0 in round-3 histogram).
__device__ __forceinline__ float xhalf(float x) {
#if defined(__has_builtin) && __has_builtin(__builtin_amdgcn_permlanex16)
    int v = __float_as_int(x);
    v = __builtin_amdgcn_permlanex16(v, v, 0x76543210, 0xfedcba98, false, false);
    return __int_as_float(v);
#else
    return __shfl_xor(x, 16, 32);
#endif
}

// ---------------------------------------------------------------------------
// Prep kernels
// ---------------------------------------------------------------------------
__global__ void cvt_f16_kernel(const float* __restrict__ s, _Float16* __restrict__ d, int n) {
    int i = blockIdx.x * blockDim.x + threadIdx.x;
    if (i < n) d[i] = (_Float16)s[i];
}

// WT[n][k] = W[k][n], 1024x1024, fp32 -> f16
__global__ void transpose_cvt_kernel(const float* __restrict__ W, _Float16* __restrict__ WT) {
    int idx = blockIdx.x * blockDim.x + threadIdx.x;
    if (idx < 1024 * 1024) {
        int n = idx >> 10, k = idx & 1023;
        WT[idx] = (_Float16)W[k * 1024 + n];
    }
}

// keybias[bh][j] = phi[b][j] - ||k_{bh,j}||^2   (the -||q||^2 row term cancels in softmax)
__global__ void keybias_kernel(const _Float16* __restrict__ qh, const float* __restrict__ phi,
                               float* __restrict__ kb) {
    int t = blockIdx.x * blockDim.x + threadIdx.x;   // bh*2048 + j
    if (t < 32 * 2048) {
        int bh = t >> 11, j = t & 2047;
        int b = bh >> 4;
        const _Float16* row = qh + (size_t)t * 64;
        float s = 0.0f;
        #pragma unroll
        for (int d = 0; d < 64; ++d) { float q = (float)row[d]; s += q * q; }
        kb[t] = phi[b * 2048 + j] - s;
    }
}

// ---------------------------------------------------------------------------
// WMMA GEMM: C(4096x1024) = A(4096x1024,f16) * B, with BT(1024x1024,f16) row-major.
// One wave computes a 32x64 strip: 2 A fragments x 4 B fragments per K-step
// (8 WMMAs per 6 fragment loads, ~100 VGPRs -> no spills).
// MODE 0: store f32 row-major (final output)
// MODE 1: store f16 head layout      qh[(b,h,tok,dh)]
// MODE 2: store f16 head-transposed  vhT[(b,h,dh,tok)]
// ---------------------------------------------------------------------------
template <int MODE>
__global__ void gemm16_kernel(const _Float16* __restrict__ A, const _Float16* __restrict__ BT,
                              float* __restrict__ outf, _Float16* __restrict__ outh) {
    int wave = threadIdx.x >> 5, lane = threadIdx.x & 31;
    int tile = blockIdx.x * 8 + wave;       // 2048 strips = 128 (M/32) x 16 (N/64)
    int tm  = tile >> 4;
    int tng = tile & 15;
    int halfsel = lane >> 4, lm = lane & 15;

    const _Float16* arow0 = A  + (size_t)(tm * 32 + lm) * 1024;       // rows m, m+16
    const _Float16* arow1 = arow0 + 16 * 1024;
    const _Float16* brow  = BT + (size_t)(tng * 64 + lm) * 1024;      // rows n,n+16,n+32,n+48
    v8f acc[2][4] = {};
    for (int kb = 0; kb < 1024; kb += 32) {
        int ko = kb + halfsel * 8;
        v16h a0 = ld16(arow0 + ko);
        v16h a1 = ld16(arow1 + ko);
        #pragma unroll
        for (int t = 0; t < 4; ++t) {
            v16h bt = ld16(brow + t * 16 * 1024 + ko);
            acc[0][t] = wmma_f16(a0, bt, acc[0][t]);
            acc[1][t] = wmma_f16(a1, bt, acc[1][t]);
        }
    }
    #pragma unroll
    for (int s = 0; s < 2; ++s) {
        #pragma unroll
        for (int t = 0; t < 4; ++t) {
            #pragma unroll
            for (int r = 0; r < 8; ++r) {
                int gm = tm * 32 + s * 16 + r + halfsel * 8;  // token row in [0,4096)
                int gn = tng * 64 + t * 16 + lm;              // output dim in [0,1024)
                float val = acc[s][t][r];
                if (MODE == 0) {
                    outf[(size_t)gm * 1024 + gn] = val;
                } else {
                    int b = gm >> 11, tok = gm & 2047;
                    int h = gn >> 6,  dh  = gn & 63;
                    if (MODE == 1)
                        outh[(((size_t)(b * 16 + h) * 2048 + tok) * 64) + dh] = (_Float16)val;
                    else
                        outh[(((size_t)(b * 16 + h) * 64 + dh) * 2048) + tok] = (_Float16)val;
                }
            }
        }
    }
}

// ---------------------------------------------------------------------------
// Flash-style attention, transposed-S formulation, 16 queries per wave
// (32 q/wave spilled to scratch in round 3 — 16 q keeps peak ~110 VGPRs).
//   S^T = K x Q^T : A = K-tile (16 keys x DH), B = Q^T (DH x 16 queries)
//   -> a softmax row (query) is a C-tile COLUMN: one lane + its half-partner.
//      Reductions: v_max3 trees in-lane + ONE v_permlanex16 (no DS, no waits).
//   O^T = V^T x P : the C-layout of S^T IS the B-fragment layout of P, so the
//      exp() results are packed per-lane with v_cvt — no LDS round trip.
//   O^T C-layout makes the epilogue 4 contiguous 16B stores per lane.
// ---------------------------------------------------------------------------
__global__ void attn_kernel(const _Float16* __restrict__ qh, const _Float16* __restrict__ vhT,
                            const float* __restrict__ kb, _Float16* __restrict__ wbuf) {
    int wave = threadIdx.x >> 5, lane = threadIdx.x & 31;
    int gw = blockIdx.x * 8 + wave;           // 4096 waves = 32 (bh) x 128 (q-tiles)
    int bh = gw >> 7, mt = gw & 127;
    int halfsel = lane >> 4, lm = lane & 15;

    const _Float16* qbase = qh  + (size_t)bh * (2048 * 64);
    const _Float16* vbase = vhT + (size_t)bh * (64 * 2048);
    const float*    kbase = kb  + (size_t)bh * 2048;

    // Q^T B-fragments for DH k-steps 0..31 / 32..63 (resident all loop)
    const _Float16* qrow = qbase + (size_t)(mt * 16 + lm) * 64;
    v16h bq0 = ld16(qrow + halfsel * 8);
    v16h bq1 = ld16(qrow + 32 + halfsel * 8);

    float mi = -1e30f, li = 0.0f;             // per-lane scalars (one query/lane)
    v8f o0 = {}, o1 = {}, o2 = {}, o3 = {};   // O^T tiles: dh 0-15,16-31,32-47,48-63

    for (int j = 0; j < 2048; j += 32) {
        // --- S^T for 32 keys (two 16-key sub-tiles; tied K == Q) ---
        const _Float16* kra = qbase + (size_t)(j + lm) * 64;
        const _Float16* krb = kra + 16 * 64;
        v8f ca = {}, cb = {};
        ca = wmma_f16(ld16(kra + halfsel * 8),      bq0, ca);
        ca = wmma_f16(ld16(kra + 32 + halfsel * 8), bq1, ca);
        cb = wmma_f16(ld16(krb + halfsel * 8),      bq0, cb);
        cb = wmma_f16(ld16(krb + 32 + halfsel * 8), bq1, cb);

        // per-key bias (phi_j - ||k_j||^2): key = j + (tile? 16:0) + halfsel*8 + r
        v8f biasa = *(const v8f*)(kbase + j + halfsel * 8);
        v8f biasb = *(const v8f*)(kbase + j + 16 + halfsel * 8);

        float sa[8], sb[8];
        float mloc = -1e30f;
        #pragma unroll
        for (int r = 0; r < 8; ++r) {
            sa[r] = 2.0f * ca[r] + biasa[r];
            sb[r] = 2.0f * cb[r] + biasb[r];
            mloc = fmaxf(mloc, fmaxf(sa[r], sb[r]));
        }
        mloc = fmaxf(mloc, xhalf(mloc));      // combine half-partners (VALU permlane)
        float mnew  = fmaxf(mi, mloc);
        float scale = __expf(mi - mnew);

        float rs = 0.0f;
        V16U pu;
        #pragma unroll
        for (int r = 0; r < 8; ++r) {
            float ea = __expf(sa[r] - mnew);
            float eb = __expf(sb[r] - mnew);
            rs += ea + eb;
            pu.v[r]     = (_Float16)ea;       // C-layout of S^T == B-layout of P
            pu.v[r + 8] = (_Float16)eb;
        }
        rs += xhalf(rs);
        li = li * scale + rs;
        mi = mnew;
        #pragma unroll
        for (int r = 0; r < 8; ++r) { o0[r] *= scale; o1[r] *= scale; o2[r] *= scale; o3[r] *= scale; }
        v16h pf = pu.v;

        // --- O^T += V^T * P  (vhT rows contiguous over keys -> A-fragments) ---
        const _Float16* vr = vbase + (size_t)lm * 2048 + j + halfsel * 8;
        o0 = wmma_f16(ld16(vr),             pf, o0);
        o1 = wmma_f16(ld16(vr + 16 * 2048), pf, o1);
        o2 = wmma_f16(ld16(vr + 32 * 2048), pf, o2);
        o3 = wmma_f16(ld16(vr + 48 * 2048), pf, o3);
    }

    // normalize and store merged-head f16 activations; O^T rows (dh) are
    // contiguous over the 8 C registers -> one 16B store per tile.
    int b = bh >> 4, h = bh & 15;
    int tok = mt * 16 + lm;                   // this lane's query token
    float inv = 1.0f / li;
    _Float16* wrow = wbuf + ((size_t)(b * 2048 + tok)) * 1024 + h * 64 + halfsel * 8;
    V8U s0, s1, s2, s3;
    #pragma unroll
    for (int r = 0; r < 8; ++r) {
        s0.e[r] = (_Float16)(o0[r] * inv);
        s1.e[r] = (_Float16)(o1[r] * inv);
        s2.e[r] = (_Float16)(o2[r] * inv);
        s3.e[r] = (_Float16)(o3[r] * inv);
    }
    *(v8h*)(wrow +  0) = s0.v;
    *(v8h*)(wrow + 16) = s1.v;
    *(v8h*)(wrow + 32) = s2.v;
    *(v8h*)(wrow + 48) = s3.v;
}

// ---------------------------------------------------------------------------
// Launch
// ---------------------------------------------------------------------------
extern "C" void kernel_launch(void* const* d_in, const int* in_sizes, int n_in,
                              void* d_out, int out_size, void* d_ws, size_t ws_size,
                              hipStream_t stream) {
    (void)in_sizes; (void)n_in; (void)out_size; (void)ws_size;
    const float* x    = (const float*)d_in[0];   // (2,2048,1024)
    const float* phi  = (const float*)d_in[1];   // (2,2048)
    const float* Wqk  = (const float*)d_in[2];   // (1024,1024)
    const float* Wv   = (const float*)d_in[3];
    const float* Wout = (const float*)d_in[4];
    float* out = (float*)d_out;                  // (2,2048,1024) f32

    char* ws = (char*)d_ws;
    _Float16* xh   = (_Float16*)(ws + 0);         //  8 MB  x f16
    _Float16* wqkT = (_Float16*)(ws + 8388608);   //  2 MB
    _Float16* wvT  = (_Float16*)(ws + 10485760);  //  2 MB
    _Float16* woT  = (_Float16*)(ws + 12582912);  //  2 MB
    _Float16* qh   = (_Float16*)(ws + 14680064);  //  8 MB  (B,H,N,DH)
    _Float16* vhT  = (_Float16*)(ws + 23068672);  //  8 MB  (B,H,DH,N)
    float*    kbp  = (float*)   (ws + 31457280);  // 256 KB (B,H,N)
    _Float16* wbuf = (_Float16*)(ws + 31719424);  //  8 MB  (B*N, D)

    cvt_f16_kernel<<<(4194304 + 255) / 256, 256, 0, stream>>>(x, xh, 4194304);
    transpose_cvt_kernel<<<4096, 256, 0, stream>>>(Wqk,  wqkT);
    transpose_cvt_kernel<<<4096, 256, 0, stream>>>(Wv,   wvT);
    transpose_cvt_kernel<<<4096, 256, 0, stream>>>(Wout, woT);

    gemm16_kernel<1><<<256, 256, 0, stream>>>(xh, wqkT, nullptr, qh);   // qk -> head layout
    gemm16_kernel<2><<<256, 256, 0, stream>>>(xh, wvT,  nullptr, vhT);  // v  -> head-T layout
    keybias_kernel<<<256, 256, 0, stream>>>(qh, phi, kbp);
    attn_kernel<<<512, 256, 0, stream>>>(qh, vhT, kbp, wbuf);
    gemm16_kernel<0><<<256, 256, 0, stream>>>(wbuf, woT, out, nullptr); // @ W_out -> f32
}
